// GIN_2946347565433
// MI455X (gfx1250) — compile-verified, matrix operand order checked by reading, not run on hardware
//
#include <hip/hip_runtime.h>

typedef __attribute__((ext_vector_type(16))) _Float16 v16h;
typedef __attribute__((ext_vector_type(8)))  _Float16 v8h;
typedef __attribute__((ext_vector_type(4)))  _Float16 v4h;
typedef __attribute__((ext_vector_type(8)))  float    v8f;

#define NODES  50000
#define EDGES  800000
#define NGRAPH 500
#define DIMH   128

// ---------------------------------------------------------------------------
// Edge message + scatter:  agg[dst] += relu(x[src] + edge_attr @ We^T + be)
// We/be cached in LDS. DIN=128: one thread owns 4 contiguous dims (float4
// gather + 4 f32 atomics). DIN=9: scalar (edge,dim) pairs.
// ---------------------------------------------------------------------------
template<int DIN>
__global__ void edge_scatter_kernel(const float* __restrict__ xin,
                                    const long long* __restrict__ src,
                                    const long long* __restrict__ dst,
                                    const float* __restrict__ eattr,
                                    const float* __restrict__ We,
                                    const float* __restrict__ be,
                                    float* __restrict__ agg, int aggStride,
                                    int nEdges)
{
    __shared__ float sWe[DIMH * 3];
    __shared__ float sBe[DIMH];
    for (int i = threadIdx.x; i < DIN * 3; i += blockDim.x) sWe[i] = We[i];
    for (int i = threadIdx.x; i < DIN;     i += blockDim.x) sBe[i] = be[i];
    __syncthreads();

    const int stride = blockDim.x * gridDim.x;
    if constexpr ((DIN & 3) == 0) {
        constexpr int DQ = DIN / 4;                 // quads per edge (32)
        const int total = nEdges * DQ;              // 25.6M, fits int
        for (int i = blockIdx.x * blockDim.x + threadIdx.x; i < total; i += stride) {
            const int e = i / DQ;                   // DQ power-of-two -> shift
            const int d = (i - e * DQ) * 4;
            const int s = (int)src[e];
            const int t = (int)dst[e];
            const float a0 = eattr[e * 3 + 0];
            const float a1 = eattr[e * 3 + 1];
            const float a2 = eattr[e * 3 + 2];
            const float4 xv = *(const float4*)&xin[(long long)s * DIN + d];
            float m0 = fmaxf(xv.x + sWe[(d+0)*3]*a0 + sWe[(d+0)*3+1]*a1 + sWe[(d+0)*3+2]*a2 + sBe[d+0], 0.0f);
            float m1 = fmaxf(xv.y + sWe[(d+1)*3]*a0 + sWe[(d+1)*3+1]*a1 + sWe[(d+1)*3+2]*a2 + sBe[d+1], 0.0f);
            float m2 = fmaxf(xv.z + sWe[(d+2)*3]*a0 + sWe[(d+2)*3+1]*a1 + sWe[(d+2)*3+2]*a2 + sBe[d+2], 0.0f);
            float m3 = fmaxf(xv.w + sWe[(d+3)*3]*a0 + sWe[(d+3)*3+1]*a1 + sWe[(d+3)*3+2]*a2 + sBe[d+3], 0.0f);
            float* ap = &agg[(long long)t * aggStride + d];
            atomicAdd(ap + 0, m0);
            atomicAdd(ap + 1, m1);
            atomicAdd(ap + 2, m2);
            atomicAdd(ap + 3, m3);
        }
    } else {
        const int total = nEdges * DIN;
        for (int i = blockIdx.x * blockDim.x + threadIdx.x; i < total; i += stride) {
            const int e = i / DIN;
            const int d = i - e * DIN;
            const int s = (int)src[e];
            const int t = (int)dst[e];
            const float a0 = eattr[e * 3 + 0];
            const float a1 = eattr[e * 3 + 1];
            const float a2 = eattr[e * 3 + 2];
            float m = xin[(long long)s * DIN + d] +
                      sWe[d*3]*a0 + sWe[d*3+1]*a1 + sWe[d*3+2]*a2 + sBe[d];
            atomicAdd(&agg[(long long)t * aggStride + d], fmaxf(m, 0.0f));
        }
    }
}

// ---------------------------------------------------------------------------
// WMMA GEMM:  C[n,128] = preop(A1 (+A2)) @ W^T   (W is [128, DIN] row-major)
//   preop   : ADD2 -> A1+A2;  BN -> normalize+ReLU (per-k mean/rstd/g/b)
//   epilogue: BIASRELU -> +bias, ReLU;  POOL -> fused graph-pool atomicAdd
// Block = 256 threads = 8 waves; block computes 16 rows x 128 cols.
// Grid must be exactly nRows/16 row-tiles (nRows % 16 == 0).
// f16 operands in LDS (+8-half row pad, 16B aligned), f32 accumulate.
// ---------------------------------------------------------------------------
template<int KPAD, int DIN, bool ADD2, bool BN, bool BIASRELU, bool POOL>
__global__ __launch_bounds__(256)
void gemm_wmma_kernel(const float* __restrict__ A1, int s1,
                      const float* __restrict__ A2, int s2,
                      const float* __restrict__ W,
                      const float* __restrict__ bnMean,
                      const float* __restrict__ bnRstd,
                      const float* __restrict__ bnG,
                      const float* __restrict__ bnB,
                      const float* __restrict__ bias,
                      float* __restrict__ C,
                      const long long* __restrict__ batch,
                      float* __restrict__ poolOut, int poolOff)
{
    constexpr int LDK = KPAD + 8;
    __shared__ _Float16 sA[16 * LDK];
    __shared__ _Float16 sB[DIMH * LDK];

    const int tid     = threadIdx.x;
    const int rowBase = blockIdx.x * 16;

    if constexpr (DIN == KPAD) {
        constexpr int KQ = KPAD / 4;               // float4 quads per row
        // W -> sB  (global_load_b128, packed v4h LDS stores)
        for (int idx = tid; idx < DIMH * KQ; idx += 256) {
            const int r = idx / KQ, k4 = idx - r * KQ;
            const float4 w = ((const float4*)(W + (long long)r * DIN))[k4];
            v4h hv = { (_Float16)w.x, (_Float16)w.y, (_Float16)w.z, (_Float16)w.w };
            *(v4h*)&sB[r * LDK + 4 * k4] = hv;
        }
        // A tile -> sA with fused add / BN+ReLU preop
        for (int idx = tid; idx < 16 * KQ; idx += 256) {
            const int m = idx / KQ, k4 = idx - m * KQ;
            const int row = rowBase + m;
            const int k = 4 * k4;
            float4 v = ((const float4*)(A1 + (long long)row * s1))[k4];
            if constexpr (ADD2) {
                const float4 u = ((const float4*)(A2 + (long long)row * s2))[k4];
                v.x += u.x; v.y += u.y; v.z += u.z; v.w += u.w;
            }
            if constexpr (BN) {
                const float4 mu = ((const float4*)bnMean)[k4];
                const float4 rs = ((const float4*)bnRstd)[k4];
                const float4 gg = ((const float4*)bnG)[k4];
                const float4 bb = ((const float4*)bnB)[k4];
                v.x = fmaxf((v.x - mu.x) * rs.x * gg.x + bb.x, 0.0f);
                v.y = fmaxf((v.y - mu.y) * rs.y * gg.y + bb.y, 0.0f);
                v.z = fmaxf((v.z - mu.z) * rs.z * gg.z + bb.z, 0.0f);
                v.w = fmaxf((v.w - mu.w) * rs.w * gg.w + bb.w, 0.0f);
            }
            v4h hv = { (_Float16)v.x, (_Float16)v.y, (_Float16)v.z, (_Float16)v.w };
            *(v4h*)&sA[m * LDK + k] = hv;
        }
    } else {
        // Scalar staging with compile-time K zero-padding (layer-0 DIN=9)
        for (int idx = tid; idx < DIMH * KPAD; idx += 256) {
            const int r = idx / KPAD, k = idx - r * KPAD;
            const float v = (k < DIN) ? W[r * DIN + k] : 0.0f;
            sB[r * LDK + k] = (_Float16)v;
        }
        for (int idx = tid; idx < 16 * KPAD; idx += 256) {
            const int m = idx / KPAD, k = idx - m * KPAD;
            const int row = rowBase + m;
            float v = 0.0f;
            if (k < DIN) {
                v = A1[(long long)row * s1 + k];
                if constexpr (ADD2) v += A2[(long long)row * s2 + k];
            }
            sA[m * LDK + k] = (_Float16)v;
        }
    }
    __syncthreads();

    const int wave    = tid >> 5;
    const int lane    = tid & 31;
    const int colBase = wave * 16;
    const int lhalf   = lane >> 4;                 // 0 or 1
    const int l16     = lane & 15;

    v8f acc = {};
    #pragma unroll
    for (int kb = 0; kb < KPAD; kb += 32) {
        const int koff = kb + lhalf * 8;
        union { v16h v; v8h h[2]; } ua, ub;
        ua.h[0] = *(const v8h*)&sA[l16 * LDK + koff];
        ua.h[1] = *(const v8h*)&sA[l16 * LDK + koff + 16];
        ub.h[0] = *(const v8h*)&sB[(colBase + l16) * LDK + koff];
        ub.h[1] = *(const v8h*)&sB[(colBase + l16) * LDK + koff + 16];
        acc = __builtin_amdgcn_wmma_f32_16x16x32_f16(
                  false, ua.v, false, ub.v, (short)0, acc, false, false);
    }

    // D layout: lane l16 = column, VGPR r = row (m = r + 8*lhalf)
    const int nCol  = colBase + l16;
    const int mBase = lhalf * 8;
    float bv = 0.0f;
    if constexpr (BIASRELU) bv = bias[nCol];       // hoisted, loop-invariant
    #pragma unroll
    for (int r = 0; r < 8; ++r) {
        const int row = rowBase + mBase + r;
        float v = acc[r];
        if constexpr (BIASRELU) v = fmaxf(v + bv, 0.0f);
        C[(long long)row * DIMH + nCol] = v;
        if constexpr (POOL) {
            const int gidx = (int)batch[row];
            atomicAdd(&poolOut[gidx * (3 * DIMH) + poolOff + nCol], v);
        }
    }
}

// ---------------------------------------------------------------------------
// BatchNorm training stats: per-column sum / sum-of-squares over N rows.
// ---------------------------------------------------------------------------
__global__ void bn_stats_kernel(const float* __restrict__ h, int n,
                                float* __restrict__ sum,
                                float* __restrict__ sumsq)
{
    const int c     = threadIdx.x & (DIMH - 1);
    const int rlane = threadIdx.x >> 7;            // 0..1
    const int rowBase = blockIdx.x * 256;
    int rowEnd = rowBase + 256; if (rowEnd > n) rowEnd = n;
    float s = 0.0f, s2 = 0.0f;
    for (int r = rowBase + rlane; r < rowEnd; r += 2) {
        const float v = h[(long long)r * DIMH + c];
        s += v; s2 += v * v;
    }
    atomicAdd(&sum[c], s);
    atomicAdd(&sumsq[c], s2);
}

__global__ void bn_finalize_kernel(const float* __restrict__ sum,
                                   const float* __restrict__ sumsq, int n,
                                   float* __restrict__ mean,
                                   float* __restrict__ rstd)
{
    const int c = threadIdx.x;
    if (c < DIMH) {
        const float inv = 1.0f / (float)n;
        const float mu  = sum[c] * inv;
        const float var = sumsq[c] * inv - mu * mu;
        mean[c] = mu;
        rstd[c] = rsqrtf(var + 1e-5f);
    }
}

// ---------------------------------------------------------------------------
extern "C" void kernel_launch(void* const* d_in, const int* in_sizes, int n_in,
                              void* d_out, int out_size, void* d_ws, size_t ws_size,
                              hipStream_t stream)
{
    (void)in_sizes; (void)n_in; (void)out_size; (void)ws_size;

    const float*     x     = (const float*)d_in[0];
    const long long* ei    = (const long long*)d_in[1];
    const long long* srcI  = ei;
    const long long* dstI  = ei + EDGES;
    const float*     eattr = (const float*)d_in[2];
    const long long* batch = (const long long*)d_in[3];
    const float* We[3], *be[3], *W1[3], *g[3], *b[3], *W2[3], *b2[3];
    for (int l = 0; l < 3; ++l) {
        const int base = 4 + 7 * l;
        We[l] = (const float*)d_in[base + 0];
        be[l] = (const float*)d_in[base + 1];
        W1[l] = (const float*)d_in[base + 2];
        g [l] = (const float*)d_in[base + 3];
        b [l] = (const float*)d_in[base + 4];
        W2[l] = (const float*)d_in[base + 5];
        b2[l] = (const float*)d_in[base + 6];
    }
    float* out = (float*)d_out;

    // Workspace: agg | h | feat | colsum | colsumsq | mean | rstd
    const size_t NF = (size_t)NODES * DIMH;
    float* agg    = (float*)d_ws;
    float* h      = agg  + NF;
    float* feat   = h    + NF;
    float* colsum = feat + NF;
    float* colsq  = colsum + DIMH;
    float* mean   = colsq  + DIMH;
    float* rstd   = mean   + DIMH;

    const int EBLK = 8192;                         // grid-stride edge blocks
    const int GBLK = NODES / 16;                   // 3125 exact row-tiles
    const int SBLK = (NODES + 255) / 256;          // BN stats blocks

    hipMemsetAsync(d_out, 0, (size_t)NGRAPH * 3 * DIMH * sizeof(float), stream);

    // ---------------- Layer 0 (din = 9, K padded to 32) ----------------
    hipMemsetAsync(agg, 0, (size_t)NODES * 16 * sizeof(float), stream);
    edge_scatter_kernel<9><<<EBLK, 256, 0, stream>>>(
        x, srcI, dstI, eattr, We[0], be[0], agg, 16, EDGES);
    hipMemsetAsync(colsum, 0, 2 * DIMH * sizeof(float), stream);
    gemm_wmma_kernel<32, 9, true, false, false, false><<<GBLK, 256, 0, stream>>>(
        agg, 16, x, 9, W1[0],
        nullptr, nullptr, nullptr, nullptr, nullptr,
        h, nullptr, nullptr, 0);
    bn_stats_kernel<<<SBLK, 256, 0, stream>>>(h, NODES, colsum, colsq);
    bn_finalize_kernel<<<1, DIMH, 0, stream>>>(colsum, colsq, NODES, mean, rstd);
    gemm_wmma_kernel<DIMH, DIMH, false, true, true, true><<<GBLK, 256, 0, stream>>>(
        h, DIMH, nullptr, 0, W2[0],
        mean, rstd, g[0], b[0], b2[0],
        feat, batch, out, 0);                      // fused pool -> cols [0,128)

    // ---------------- Layers 1, 2 (din = 128) ----------------
    for (int l = 1; l < 3; ++l) {
        hipMemsetAsync(agg, 0, NF * sizeof(float), stream);
        edge_scatter_kernel<DIMH><<<EBLK, 256, 0, stream>>>(
            feat, srcI, dstI, eattr, We[l], be[l], agg, DIMH, EDGES);
        hipMemsetAsync(colsum, 0, 2 * DIMH * sizeof(float), stream);
        gemm_wmma_kernel<DIMH, DIMH, true, false, false, false><<<GBLK, 256, 0, stream>>>(
            agg, DIMH, feat, DIMH, W1[l],
            nullptr, nullptr, nullptr, nullptr, nullptr,
            h, nullptr, nullptr, 0);
        bn_stats_kernel<<<SBLK, 256, 0, stream>>>(h, NODES, colsum, colsq);
        bn_finalize_kernel<<<1, DIMH, 0, stream>>>(colsum, colsq, NODES, mean, rstd);
        // GEMM2 reads only h; safe to overwrite feat in place. Fused pool.
        gemm_wmma_kernel<DIMH, DIMH, false, true, true, true><<<GBLK, 256, 0, stream>>>(
            h, DIMH, nullptr, 0, W2[l],
            mean, rstd, g[l], b[l], b2[l],
            feat, batch, out, DIMH * l);
    }
}
